// _MSA1_58780922413931
// MI455X (gfx1250) — matrix operand order repaired and checked by reading, not compiled, not run on hardware
//
#include <hip/hip_runtime.h>
#include <stdint.h>

typedef __attribute__((ext_vector_type(16))) __bf16 v16bf;
typedef __attribute__((ext_vector_type(8)))  float  v8f;
typedef __attribute__((ext_vector_type(4))) unsigned int v4u;
typedef __attribute__((ext_vector_type(8))) int v8i;
typedef __attribute__((ext_vector_type(4))) int v4i;

#if defined(__has_builtin)
#  if __has_builtin(__builtin_amdgcn_tensor_load_to_lds)
#    define HAVE_TDM 1
#  else
#    define HAVE_TDM 0
#  endif
#  if __has_builtin(__builtin_amdgcn_s_wait_tensorcnt)
#    define WAIT_TENSOR() __builtin_amdgcn_s_wait_tensorcnt(0)
#  else
#    define WAIT_TENSOR() asm volatile("s_wait_tensorcnt 0x0" ::: "memory")
#  endif
#else
#  define HAVE_TDM 0
#  define WAIT_TENSOR()
#endif

namespace {
constexpr int B_    = 8;
constexpr int NHID_ = 512;
constexpr int L_    = 2048;
constexpr int AL_   = 4;
constexpr int NHEAD_= 8;
constexpr int HD_   = 64;
constexpr int D_    = 512;          // NHEAD * HD
constexpr int M1    = B_ * L_;      // 16384 rows (main sequence)
constexpr int M2    = B_ * AL_ * L_;// 65536 rows (axial)
constexpr int KT    = NHID_ / 32;   // 16 K-tiles of 32
constexpr float SCALE_ = 0.125f;    // 1/sqrt(64)
constexpr int NWIN  = 3 + AL_;      // 7 softmax entries
}

// ---------------- bf16 helpers ----------------
__device__ __forceinline__ unsigned short f2bf(float f) {
  unsigned u = __builtin_bit_cast(unsigned, f);
  u = (u + 0x7FFFu + ((u >> 16) & 1u)) >> 16;
  return (unsigned short)u;
}
__device__ __forceinline__ float bf2f(unsigned short h) {
  unsigned u = ((unsigned)h) << 16;
  return __builtin_bit_cast(float, u);
}

// Packed WMMA A-fragment index for element (row m, col k).
// 16x32 bf16 A tile, ISA layout: lane = r + 16*half, 16 values/lane.
__device__ __forceinline__ size_t apack_idx(int m, int k) {
  int mT   = m >> 4, r  = m & 15;
  int kt   = k >> 5, kk = k & 31;
  int half = (kk >> 3) & 1;
  int slot = (kk & 7) | (((kk >> 4) & 1) << 3);
  int lane = r | (half << 4);
  return ((((size_t)mT * KT) + kt) * 32 + lane) * 16 + slot;
}

// ---------------- pack activations (x + ax) -> bf16 A fragments ----------------
__global__ __launch_bounds__(256) void pack_a_kernel(
    const float* __restrict__ x, const float* __restrict__ ax,
    unsigned short* __restrict__ Apk) {
  const int m = blockIdx.x * 256 + threadIdx.x;   // 0 .. M1+M2-1
  const int c = blockIdx.y;                       // 0 .. 511
  float val;
  if (m < M1) {
    const int b = m >> 11, l = m & 2047;
    val = x[(((size_t)b * NHID_ + c) << 11) + l];
  } else {
    const int mm = m - M1;
    const int b = mm >> 13;
    const int rem = mm & 8191;
    const int a = rem >> 11, l = rem & 2047;
    val = ax[((((size_t)b * NHID_ + c) * AL_ + a) << 11) + l];
  }
  Apk[apack_idx(m, c)] = f2bf(val);
}

// ---------------- pack weights -> three concatenated bf16 B panels ----------------
// panel 0: [Wq|Wk|Wv]  N=1536 (NTT=96)   panel 1: [Wk|Wv] N=1024 (NTT=64)
// panel 2: [Wo]        N=512  (NTT=32)
__global__ __launch_bounds__(256) void pack_w_kernel(
    const float* __restrict__ Wq, const float* __restrict__ Wk,
    const float* __restrict__ Wv, const float* __restrict__ Wo,
    unsigned short* __restrict__ Wqkv, unsigned short* __restrict__ Wkvax,
    unsigned short* __restrict__ Wop) {
  const int gid  = blockIdx.x * 256 + threadIdx.x;   // 6*512*512 threads
  const int which= gid >> 18;
  const int k    = (gid >> 9) & 511;
  const int n    = gid & 511;
  const float* src; unsigned short* dst; int ntt, noff;
  switch (which) {
    case 0: src = Wq; dst = Wqkv;  ntt = 96; noff = 0;    break;
    case 1: src = Wk; dst = Wqkv;  ntt = 96; noff = 512;  break;
    case 2: src = Wv; dst = Wqkv;  ntt = 96; noff = 1024; break;
    case 3: src = Wk; dst = Wkvax; ntt = 64; noff = 0;    break;
    case 4: src = Wv; dst = Wkvax; ntt = 64; noff = 512;  break;
    default:src = Wo; dst = Wop;   ntt = 32; noff = 0;    break;
  }
  const float val = src[(size_t)k * 512 + n];
  const int ng = noff + n;
  const int kt = k >> 5, kk = k & 31;
  const int nt = ng >> 4, c = ng & 15;
  const int half = (kk >> 3) & 1, slot = (kk & 7) | (((kk >> 4) & 1) << 3);
  const int lane = c | (half << 4);
  dst[(((size_t)kt * ntt + nt) * 32 + lane) * 16 + slot] = f2bf(val);
}

// ---------------- WMMA GEMM with TDM-staged LDS weight panel ----------------
// MODE 0: fused QKV   (NTT=96),  bf16 out -> Qt/Kt/Vt   [B, D, L]
// MODE 1: fused ax KV (NTT=64),  bf16 out -> AKt/AVt    [B, D, AL, L]
// MODE 2: output proj (NTT=32),  f32  out -> d_out      [B, NHID, L]
template <int MODE>
__global__ __launch_bounds__(256) void gemm_bf16_kernel(
    const unsigned short* __restrict__ Apk,
    const unsigned short* __restrict__ Wpk,
    const float* __restrict__ bias0, const float* __restrict__ bias1,
    const float* __restrict__ bias2,
    void* __restrict__ out0, void* __restrict__ out1, void* __restrict__ out2) {
  constexpr int NTT = (MODE == 0) ? 96 : (MODE == 1) ? 64 : 32;
  __shared__ __align__(16) unsigned short smem[KT * 4 * 512];   // 64 KB B panel

  const int lane  = threadIdx.x & 31;
  const int wave  = threadIdx.x >> 5;
  const int mTile = blockIdx.y * 8 + wave;   // 16 rows per wave
  const int ntB   = blockIdx.x * 4;          // 64 cols per block
  const unsigned short* Wbase = Wpk + (size_t)ntB * 512;

#if HAVE_TDM
  if (threadIdx.x == 0) {
    // Tensor DMA descriptor: 2D tile, 16 rows x 1024 dwords, data_size = 4B,
    // row stride NTT*256 dwords, destination = smem (LDS offset in low 32 bits
    // of the generic pointer).
    const uint64_t ga = (uint64_t)(uintptr_t)Wbase;
    const uint32_t ldsa = (uint32_t)(uintptr_t)(&smem[0]);
    v4u g0; v8i g1; v4i gz0, gz1; v8i gz2;
    g0[0] = 1u;                                  // count=1, user descriptor
    g0[1] = ldsa;                                // lds_addr
    g0[2] = (uint32_t)ga;                        // global_addr[31:0]
    g0[3] = (uint32_t)((ga >> 32) & 0x01FFFFFFu) | 0x80000000u;  // addr[56:32] | type=2
    g1[0] = 0x00020000;                          // data_size = 2 (4 bytes)
    g1[1] = (NTT * 256) << 16;                   // tensor_dim0 (row length, dwords)
    g1[2] = 16 << 16;                            // tensor_dim1 = 16 rows
    g1[3] = 1024 << 16;                          // tile_dim0 = 1024 dwords (4KB)
    g1[4] = 16;                                  // tile_dim1 = 16, tile_dim2 = 0
    g1[5] = NTT * 256;                           // tensor_dim0_stride (dwords)
    g1[6] = 0; g1[7] = 0;
    gz0[0]=gz0[1]=gz0[2]=gz0[3]=0;
    gz1[0]=gz1[1]=gz1[2]=gz1[3]=0;
    gz2[0]=gz2[1]=gz2[2]=gz2[3]=gz2[4]=gz2[5]=gz2[6]=gz2[7]=0;
    __builtin_amdgcn_tensor_load_to_lds(g0, g1, gz0, gz1, gz2, 0);
    WAIT_TENSOR();
  }
#else
  // cooperative fallback: 16 rows x 4096B, b128 per thread-chunk
  for (int t = threadIdx.x; t < KT * 256; t += 256) {
    const int row = t >> 8, chunk = t & 255;
    *(uint4*)(smem + (size_t)row * 2048 + chunk * 8) =
        *(const uint4*)(Wbase + (size_t)row * (NTT * 512) + chunk * 8);
  }
#endif
  __syncthreads();

  const v16bf* __restrict__ Av = reinterpret_cast<const v16bf*>(Apk);
  const v16bf* __restrict__ Bs = reinterpret_cast<const v16bf*>(smem);
  const size_t abase = (size_t)mTile * KT * 32 + lane;

  v8f acc[4] = {v8f{}, v8f{}, v8f{}, v8f{}};
  v16bf a = Av[abase];                     // prefetch kt=0
#pragma unroll
  for (int kt = 0; kt < KT; ++kt) {
    v16bf an = a;
    if (kt + 1 < KT) an = Av[abase + (size_t)(kt + 1) * 32];   // prefetch next
#pragma unroll
    for (int j = 0; j < 4; ++j) {
      const v16bf b = Bs[(kt * 4 + j) * 32 + lane];            // ds_load from panel
      acc[j] = __builtin_amdgcn_wmma_f32_16x16x32_bf16(
          false, a, false, b, (short)0, acc[j], false, false);
    }
    a = an;
  }

  // f32 C/D layout: lane 0-15 -> M = i, lane 16-31 -> M = 8 + i ; N = lane & 15
  const int rhalf = lane >> 4;
  const int col   = lane & 15;
#pragma unroll
  for (int j = 0; j < 4; ++j) {
    const int ng = (ntB + j) * 16 + col;
    const float* bp; void* op; int c;
    if (MODE == 0) {
      const int which = ng >> 9; c = ng & 511;
      bp = (which == 0) ? bias0 : (which == 1) ? bias1 : bias2;
      op = (which == 0) ? out0 : (which == 1) ? out1 : out2;
    } else if (MODE == 1) {
      const int which = ng >> 9; c = ng & 511;
      bp = (which == 0) ? bias0 : bias1;
      op = (which == 0) ? out0 : out1;
    } else {
      c = ng; bp = bias0; op = out0;
    }
    const float bb = bp[c];
#pragma unroll
    for (int i = 0; i < 8; ++i) {
      const int m = mTile * 16 + rhalf * 8 + i;
      const float val = acc[j][i] + bb;
      if (MODE == 0) {
        const int b = m >> 11, l = m & 2047;
        ((unsigned short*)op)[(((size_t)b * D_ + c) << 11) + l] = f2bf(val);
      } else if (MODE == 1) {
        const int b = m >> 13, rem = m & 8191;
        const int a2 = rem >> 11, l = rem & 2047;
        ((unsigned short*)op)[((((size_t)b * D_ + c) * AL_ + a2) << 11) + l] = f2bf(val);
      } else {
        const int b = m >> 11, l = m & 2047;
        ((float*)op)[(((size_t)b * NHID_ + c) << 11) + l] = val;
      }
    }
  }
}

// ---------------- attention: 7-entry softmax per (b, head, l) ----------------
__global__ __launch_bounds__(256) void attn_kernel(
    const unsigned short* __restrict__ Qt, const unsigned short* __restrict__ Kt,
    const unsigned short* __restrict__ Vt, const unsigned short* __restrict__ AKt,
    const unsigned short* __restrict__ AVt, unsigned short* __restrict__ AttPk) {
  const int l = blockIdx.x * 256 + threadIdx.x;
  const int n = blockIdx.y;
  const int b = blockIdx.z;
  const int ndB = n * HD_;

  float sc[NWIN];
#pragma unroll
  for (int w = 0; w < NWIN; ++w) sc[w] = 0.0f;

  const size_t rowQ = (((size_t)b * D_ + ndB) << 11);
#pragma unroll 4
  for (int d = 0; d < HD_; ++d) {
    const size_t kr = rowQ + ((size_t)d << 11);
    const float qd = bf2f(Qt[kr + l]);
#pragma unroll
    for (int w = 0; w < 3; ++w) {
      const int p = l + w - 1;                  // SAME zero padding
      if (p >= 0 && p < L_) sc[w] += qd * bf2f(Kt[kr + p]);
    }
    const size_t ar = ((((size_t)b * D_ + ndB + d) * AL_) << 11);
#pragma unroll
    for (int a = 0; a < AL_; ++a)
      sc[3 + a] += qd * bf2f(AKt[ar + ((size_t)a << 11) + l]);
  }

  float mx = sc[0] * SCALE_;
#pragma unroll
  for (int w = 0; w < NWIN; ++w) { sc[w] *= SCALE_; mx = fmaxf(mx, sc[w]); }
  float sum = 0.0f;
  float alpha[NWIN];
#pragma unroll
  for (int w = 0; w < NWIN; ++w) { alpha[w] = __expf(sc[w] - mx); sum += alpha[w]; }
  const float inv = 1.0f / sum;
#pragma unroll
  for (int w = 0; w < NWIN; ++w) alpha[w] *= inv;

  const int m = b * L_ + l;
#pragma unroll 4
  for (int d = 0; d < HD_; ++d) {
    const size_t kr = rowQ + ((size_t)d << 11);
    float o = 0.0f;
#pragma unroll
    for (int w = 0; w < 3; ++w) {
      const int p = l + w - 1;
      if (p >= 0 && p < L_) o += alpha[w] * bf2f(Vt[kr + p]);
    }
    const size_t ar = ((((size_t)b * D_ + ndB + d) * AL_) << 11);
#pragma unroll
    for (int a = 0; a < AL_; ++a)
      o += alpha[3 + a] * bf2f(AVt[ar + ((size_t)a << 11) + l]);
    AttPk[apack_idx(m, ndB + d)] = f2bf(o);
  }
}

// ---------------- host-side orchestration ----------------
extern "C" void kernel_launch(void* const* d_in, const int* in_sizes, int n_in,
                              void* d_out, int out_size, void* d_ws, size_t ws_size,
                              hipStream_t stream) {
  (void)in_sizes; (void)n_in; (void)out_size; (void)ws_size;
  const float* x  = (const float*)d_in[0];
  const float* ax = (const float*)d_in[1];
  const float* Wq = (const float*)d_in[2];
  const float* bq = (const float*)d_in[3];
  const float* Wk = (const float*)d_in[4];
  const float* bk = (const float*)d_in[5];
  const float* Wv = (const float*)d_in[6];
  const float* bv = (const float*)d_in[7];
  const float* Wo = (const float*)d_in[8];
  const float* bo = (const float*)d_in[9];

  char* ws = (char*)d_ws;
  size_t off = 0;
  auto carve = [&](size_t bytes) {
    char* p = ws + off;
    off += (bytes + 255) & ~(size_t)255;
    return p;
  };
  unsigned short* Apk   = (unsigned short*)carve((size_t)(M1 + M2) * 512 * 2);
  unsigned short* Wqkv  = (unsigned short*)carve((size_t)3 * 512 * 512 * 2);
  unsigned short* Wkvax = (unsigned short*)carve((size_t)2 * 512 * 512 * 2);
  unsigned short* Wop   = (unsigned short*)carve((size_t)1 * 512 * 512 * 2);
  unsigned short* Qt    = (unsigned short*)carve((size_t)B_ * D_ * L_ * 2);
  unsigned short* Kt    = (unsigned short*)carve((size_t)B_ * D_ * L_ * 2);
  unsigned short* Vt    = (unsigned short*)carve((size_t)B_ * D_ * L_ * 2);
  unsigned short* AKt   = (unsigned short*)carve((size_t)B_ * D_ * AL_ * L_ * 2);
  unsigned short* AVt   = (unsigned short*)carve((size_t)B_ * D_ * AL_ * L_ * 2);
  unsigned short* AttPk = (unsigned short*)carve((size_t)M1 * 512 * 2);

  // 1) pack activations + weights into WMMA fragment layouts (fp32 -> bf16)
  pack_a_kernel<<<dim3((M1 + M2) / 256, NHID_), 256, 0, stream>>>(x, ax, Apk);
  pack_w_kernel<<<(6 * 512 * 512) / 256, 256, 0, stream>>>(Wq, Wk, Wv, Wo,
                                                           Wqkv, Wkvax, Wop);

  // 2) fused projection GEMMs (bf16 WMMA, f32 accumulate, TDM-staged weights)
  gemm_bf16_kernel<0><<<dim3(1536 / 64, M1 / 128), 256, 0, stream>>>(
      Apk, Wqkv, bq, bk, bv, Qt, Kt, Vt);
  const unsigned short* ApkAx = Apk + (size_t)M1 * 512;
  gemm_bf16_kernel<1><<<dim3(1024 / 64, M2 / 128), 256, 0, stream>>>(
      ApkAx, Wkvax, bk, bv, nullptr, AKt, AVt, nullptr);

  // 3) windowed + axial softmax attention, result emitted as packed-A bf16
  attn_kernel<<<dim3(L_ / 256, NHEAD_, B_), 256, 0, stream>>>(Qt, Kt, Vt, AKt, AVt,
                                                              AttPk);

  // 4) output projection, transposed store straight into d_out [B, NHID, L, 1]
  gemm_bf16_kernel<2><<<dim3(512 / 64, M1 / 128), 256, 0, stream>>>(
      AttPk, Wop, bo, nullptr, nullptr, d_out, nullptr, nullptr);
}
